// GraphAttentionLayer_47708496724107
// MI455X (gfx1250) — compile-verified, matrix operand order checked by reading, not run on hardware
//
#include <hip/hip_runtime.h>
#include <hip/hip_bf16.h>

// GAT layer: n=4096, in_dim=256, HEADS=4, OUT_DIM=64, NEG_SLOPE=0.2
#define N_NODES 4096
#define IN_DIM  256
#define NH      4
#define OD      64
#define HID     256   // NH*OD
#define SLOPE   0.2f

typedef __attribute__((ext_vector_type(16))) _Float16 v16h;
typedef __attribute__((ext_vector_type(8)))  _Float16 h8;
typedef __attribute__((ext_vector_type(8)))  float    v8f;

__device__ __forceinline__ float lrelu(float v) {
    return v >= 0.0f ? v : SLOPE * v;
}

// ---------------------------------------------------------------------------
// Kernel 1: h = x @ W  (4096x256x256), f16 WMMA, f32 accumulate.
// One wave per 16x16 output tile. Output stored TRANSPOSED as f16:
// hT[col][row], col in [0,256), row in [0,4096) -> B-fragments of the
// aggregation GEMM become contiguous 16B loads per lane.
// ---------------------------------------------------------------------------
__global__ void gat_gemm_h(const float* __restrict__ x,
                           const float* __restrict__ W,
                           _Float16* __restrict__ hT) {
    const int wid  = (blockIdx.x * blockDim.x + threadIdx.x) >> 5;
    const int lane = threadIdx.x & 31;
    const int i_tile = wid >> 4;   // 0..255  (rows of x)
    const int n_tile = wid & 15;   // 0..15   (cols of h)
    const int i0 = i_tile * 16, n0 = n_tile * 16;
    const int r  = lane & 15;      // sub-row / sub-col
    const int hi = lane >> 4;      // half-wave selector

    v8f c = {};
    const float* xrow = x + (size_t)(i0 + r) * IN_DIM;

    for (int k0 = 0; k0 < IN_DIM; k0 += 32) {
        const int kb1 = k0 + hi * 8;        // K chunk {0..7} or {8..15}
        const int kb2 = k0 + 16 + hi * 8;   // K chunk {16..23} or {24..31}
        v16h a, b;
#pragma unroll
        for (int t = 0; t < 8; ++t) {
            a[t]     = (_Float16)xrow[kb1 + t];
            a[8 + t] = (_Float16)xrow[kb2 + t];
            b[t]     = (_Float16)W[(size_t)(kb1 + t) * HID + n0 + r];
            b[8 + t] = (_Float16)W[(size_t)(kb2 + t) * HID + n0 + r];
        }
        c = __builtin_amdgcn_wmma_f32_16x16x32_f16(false, a, false, b,
                                                   (short)0, c, false, false);
    }

    // C layout: VGPR v -> row = i0 + hi*8 + v, col = n0 + r.
    _Float16* dst = hT + (size_t)(n0 + r) * N_NODES + i0 + hi * 8;
#pragma unroll
    for (int v = 0; v < 8; ++v) dst[v] = (_Float16)c[v];
}

// ---------------------------------------------------------------------------
// Kernel 2: e_src[n][h] = sum_d h[n,h,d]*att[h][d]
//           e_dstT[h][n] = sum_d h[n,h,d]*att[h][64+d]   (transposed store)
// One thread per node; loads of hT are coalesced across lanes.
// ---------------------------------------------------------------------------
__global__ void gat_edge_logits(const _Float16* __restrict__ hT,
                                const float* __restrict__ att,
                                float* __restrict__ e_src,
                                float* __restrict__ e_dstT) {
    const int n = blockIdx.x * blockDim.x + threadIdx.x;
    if (n >= N_NODES) return;
#pragma unroll
    for (int h = 0; h < NH; ++h) {
        float s = 0.0f, d = 0.0f;
        for (int dd = 0; dd < OD; ++dd) {
            const float hv = (float)hT[(size_t)(h * OD + dd) * N_NODES + n];
            s += hv * att[h * 2 * OD + dd];
            d += hv * att[h * 2 * OD + OD + dd];
        }
        e_src[n * NH + h]        = s;
        e_dstT[h * N_NODES + n]  = d;
    }
}

// ---------------------------------------------------------------------------
// Kernel 3: per-row softmax stats. Monotonicity of leaky-relu:
//   m[i][h] = lrelu(e_src[i][h] + max_{j:adj>0} e_dstT[h][j])
//   Z[i][h] = sum_{j:adj>0} exp(lrelu(e_src+e_dst) - m)
// One wave per row; lanes stride the 4096-wide adj row (coalesced, L2-hot).
// ---------------------------------------------------------------------------
__global__ void gat_softmax_stats(const int* __restrict__ adj,
                                  const float* __restrict__ e_src,
                                  const float* __restrict__ e_dstT,
                                  float* __restrict__ m_buf,
                                  float* __restrict__ z_buf) {
    const int lane = threadIdx.x & 31;
    const int i = blockIdx.x * (blockDim.x >> 5) + (threadIdx.x >> 5);
    const int* arow = adj + (size_t)i * N_NODES;
    const float NEG_INF = -__builtin_inff();

    float maxd[NH] = {NEG_INF, NEG_INF, NEG_INF, NEG_INF};
    for (int j = lane; j < N_NODES; j += 32) {
        if (arow[j] > 0) {
#pragma unroll
            for (int h = 0; h < NH; ++h)
                maxd[h] = fmaxf(maxd[h], e_dstT[h * N_NODES + j]);
        }
    }
#pragma unroll
    for (int off = 16; off > 0; off >>= 1)
#pragma unroll
        for (int h = 0; h < NH; ++h)
            maxd[h] = fmaxf(maxd[h], __shfl_xor(maxd[h], off, 32));

    float es[NH], m[NH], z[NH] = {0.f, 0.f, 0.f, 0.f};
#pragma unroll
    for (int h = 0; h < NH; ++h) {
        es[h] = e_src[i * NH + h];
        m[h]  = lrelu(es[h] + maxd[h]);   // -inf stays -inf if no neighbors
    }
    for (int j = lane; j < N_NODES; j += 32) {
        if (arow[j] > 0) {
#pragma unroll
            for (int h = 0; h < NH; ++h) {
                const float sc = lrelu(es[h] + e_dstT[h * N_NODES + j]);
                z[h] += __expf(sc - m[h]);
            }
        }
    }
#pragma unroll
    for (int off = 16; off > 0; off >>= 1)
#pragma unroll
        for (int h = 0; h < NH; ++h)
            z[h] += __shfl_xor(z[h], off, 32);

    if (lane == 0) {
#pragma unroll
        for (int h = 0; h < NH; ++h) {
            m_buf[i * NH + h] = m[h];
            z_buf[i * NH + h] = z[h];
        }
    }
}

// ---------------------------------------------------------------------------
// Kernel 4: aggregation GEMM  out[i, h*64+d] = (1/Z_i) * sum_j P[i,j]*h[j,h,d]
// with P[i,j] = adj[i,j] ? exp(lrelu(e_src_i+e_dst_j) - m_i) : 0  (<= 1, f16-safe)
// One wave per (16-row i-tile, head): 4 accumulator tiles (16x16 each),
// K-loop over j in chunks of 32. A-fragment (P) built on the fly from
// aligned int4 adj loads + v_exp_f32; B-fragment from transposed hT (two
// contiguous b128 loads per lane). ~8.6 GFLOP total on v_wmma_f32_16x16x32_f16.
// ---------------------------------------------------------------------------
__global__ void gat_aggregate(const int* __restrict__ adj,
                              const _Float16* __restrict__ hT,
                              const float* __restrict__ e_src,
                              const float* __restrict__ e_dstT,
                              const float* __restrict__ m_buf,
                              const float* __restrict__ z_buf,
                              float* __restrict__ out) {
    const int wid  = (blockIdx.x * blockDim.x + threadIdx.x) >> 5;
    const int lane = threadIdx.x & 31;
    const int i_tile = wid >> 2;   // 0..255
    const int head   = wid & 3;    // 0..3
    const int i0 = i_tile * 16;
    const int r  = lane & 15;
    const int hi = lane >> 4;

    const int   row_a = i0 + r;                       // A-fragment row
    const float es = e_src[row_a * NH + head];
    const float m  = m_buf[row_a * NH + head];
    const int*   arow = adj    + (size_t)row_a * N_NODES;
    const float* drow = e_dstT + (size_t)head  * N_NODES;

    v8f acc0 = {}, acc1 = {}, acc2 = {}, acc3 = {};

    for (int j0 = 0; j0 < N_NODES; j0 += 32) {
        const int b1 = j0 + hi * 8;        // K chunk for this half-wave
        const int b2 = j0 + 16 + hi * 8;
        __builtin_prefetch(arow + j0 + 64, 0, 1);   // -> global_prefetch_b8

        // adj mask: two aligned 32B runs per lane
        const int4 q0 = *(const int4*)(arow + b1);
        const int4 q1 = *(const int4*)(arow + b1 + 4);
        const int4 q2 = *(const int4*)(arow + b2);
        const int4 q3 = *(const int4*)(arow + b2 + 4);
        const int av[16] = {q0.x, q0.y, q0.z, q0.w, q1.x, q1.y, q1.z, q1.w,
                            q2.x, q2.y, q2.z, q2.w, q3.x, q3.y, q3.z, q3.w};
        const float4 d0 = *(const float4*)(drow + b1);
        const float4 d1 = *(const float4*)(drow + b1 + 4);
        const float4 d2 = *(const float4*)(drow + b2);
        const float4 d3 = *(const float4*)(drow + b2 + 4);
        const float dv[16] = {d0.x, d0.y, d0.z, d0.w, d1.x, d1.y, d1.z, d1.w,
                              d2.x, d2.y, d2.z, d2.w, d3.x, d3.y, d3.z, d3.w};

        v16h a;
#pragma unroll
        for (int t = 0; t < 16; ++t) {
            const float sc = lrelu(es + dv[t]);
            const float p  = (av[t] > 0) ? __expf(sc - m) : 0.0f;
            a[t] = (_Float16)p;
        }

#pragma unroll
        for (int nt = 0; nt < 4; ++nt) {
            const _Float16* hcol = hT + (size_t)(head * OD + nt * 16 + r) * N_NODES;
            const h8 blo = *(const h8*)(hcol + b1);   // contiguous b128
            const h8 bhi = *(const h8*)(hcol + b2);   // contiguous b128
            v16h b;
#pragma unroll
            for (int t = 0; t < 8; ++t) { b[t] = blo[t]; b[8 + t] = bhi[t]; }
            switch (nt) {
            case 0: acc0 = __builtin_amdgcn_wmma_f32_16x16x32_f16(false, a, false, b, (short)0, acc0, false, false); break;
            case 1: acc1 = __builtin_amdgcn_wmma_f32_16x16x32_f16(false, a, false, b, (short)0, acc1, false, false); break;
            case 2: acc2 = __builtin_amdgcn_wmma_f32_16x16x32_f16(false, a, false, b, (short)0, acc2, false, false); break;
            case 3: acc3 = __builtin_amdgcn_wmma_f32_16x16x32_f16(false, a, false, b, (short)0, acc3, false, false); break;
            }
        }
    }

    // Epilogue: scale by 1/Z and store. C elem v: row = i0 + hi*8 + v, col = n0 + r.
    float zinv[8];
#pragma unroll
    for (int v = 0; v < 8; ++v) {
        const float z = z_buf[(i0 + hi * 8 + v) * NH + head];
        zinv[v] = (z > 0.0f) ? 1.0f / z : 0.0f;
    }
#pragma unroll
    for (int v = 0; v < 8; ++v) {
        float* orow = out + (size_t)(i0 + hi * 8 + v) * HID + head * OD + r;
        orow[0]  = acc0[v] * zinv[v];
        orow[16] = acc1[v] * zinv[v];
        orow[32] = acc2[v] * zinv[v];
        orow[48] = acc3[v] * zinv[v];
    }
}

// ---------------------------------------------------------------------------
extern "C" void kernel_launch(void* const* d_in, const int* in_sizes, int n_in,
                              void* d_out, int out_size, void* d_ws, size_t ws_size,
                              hipStream_t stream) {
    const float* x   = (const float*)d_in[0];   // 4096 x 256
    const int*   adj = (const int*)  d_in[1];   // 4096 x 4096
    const float* W   = (const float*)d_in[2];   // 256 x 256
    const float* att = (const float*)d_in[3];   // 4 x 128
    float* out = (float*)d_out;                 // 4096 x 256

    // Workspace layout
    char* ws = (char*)d_ws;
    _Float16* hT     = (_Float16*)(ws);                       // 256*4096*2 = 2 MB
    float*    e_src  = (float*)(ws + 2 * 1024 * 1024);        // 64 KB
    float*    e_dstT = (float*)(ws + 2 * 1024 * 1024 + 64 * 1024);
    float*    m_buf  = (float*)(ws + 2 * 1024 * 1024 + 128 * 1024);
    float*    z_buf  = (float*)(ws + 2 * 1024 * 1024 + 192 * 1024);

    // 1) h = x@W (4096 waves = 512 blocks of 8 waves)
    gat_gemm_h<<<512, 256, 0, stream>>>(x, W, hT);
    // 2) attention logits per node
    gat_edge_logits<<<16, 256, 0, stream>>>(hT, att, e_src, e_dstT);
    // 3) per-row masked softmax stats (one wave per row)
    gat_softmax_stats<<<512, 256, 0, stream>>>(adj, e_src, e_dstT, m_buf, z_buf);
    // 4) aggregation GEMM (256 i-tiles x 4 heads = 1024 waves)
    gat_aggregate<<<128, 256, 0, stream>>>(adj, hT, e_src, e_dstT, m_buf, z_buf, out);
}
